// kernel_generated_1_62904091017251
// MI455X (gfx1250) — matrix operand order repaired and checked by reading, not compiled, not run on hardware
//
#include <hip/hip_runtime.h>

// Problem constants from the reference
#define M_  1024
#define J_  64
#define L_  28
#define H_  28
#define I_  128
#define JK_ 192          // contraction length J*K = 64*3
#define AROW 208         // padded LDS row stride (halves); 416B = 16B aligned

typedef __attribute__((ext_vector_type(16))) _Float16 v16h;
typedef __attribute__((ext_vector_type(8)))  _Float16 v8h;
typedef __attribute__((ext_vector_type(8)))  float    v8f;
typedef __attribute__((ext_vector_type(4)))  float    v4f;
typedef __attribute__((ext_vector_type(4)))  int      v4i;

#if defined(__has_builtin)
#  if __has_builtin(__builtin_amdgcn_global_load_async_to_lds_b128)
#    define HAS_ASYNC_LDS 1
#  endif
#endif

typedef __attribute__((address_space(1))) v4i as1_v4i;   // global
typedef __attribute__((address_space(3))) v4i as3_v4i;   // LDS

// Stage x[m, :, l, :] (64 rows x 28 f32) into an LDS buffer.
// Async path: 448 16-byte chunks via GLOBAL_LOAD_ASYNC_TO_LDS_B128 (ASYNCcnt).
__device__ __forceinline__ void stage_slice(const float* __restrict__ xm, int l,
                                            float* buf, int tid)
{
#ifdef HAS_ASYNC_LDS
    #pragma unroll
    for (int it = 0; it < 2; ++it) {
        const int c = tid + it * 256;
        if (c < 448) {                       // chunk c: row j=c/7, col off=4*(c%7)
            const int j   = c / 7;
            const int off = (c - j * 7) * 4;
            float* g = const_cast<float*>(xm + (j * L_ + l) * H_ + off); // 16B aligned
            __builtin_amdgcn_global_load_async_to_lds_b128(
                (as1_v4i*)g, (as3_v4i*)(buf + c * 4), 0, 0);
        }
    }
#else
    for (int f = tid; f < J_ * H_; f += 256) {
        const int j = f / H_, n = f - j * H_;
        buf[f] = xm[(j * L_ + l) * H_ + n];
    }
#endif
}

__device__ __forceinline__ void wait_async_all()
{
#ifdef HAS_ASYNC_LDS
#  if __has_builtin(__builtin_amdgcn_s_wait_asynccnt)
    __builtin_amdgcn_s_wait_asynccnt(0);
#  else
    asm volatile("s_wait_asynccnt 0x0" ::: "memory");
#  endif
#endif
}

// out = w10*cur + w11*prev for this wave's i-column, aligned float4 stores.
// c0/p0: rows n = 8*half + v ; c1/p1: rows n = 16 + 8*half + v (skip n>=28).
__device__ __forceinline__ void emit_slice(float* __restrict__ ob, int half,
                                           v8f c0, v8f p0, v8f c1, v8f p1,
                                           float w10, float w11)
{
    const int nb0 = 8 * half;
    v4f a = { c0[0]*w10 + p0[0]*w11, c0[1]*w10 + p0[1]*w11,
              c0[2]*w10 + p0[2]*w11, c0[3]*w10 + p0[3]*w11 };
    v4f b = { c0[4]*w10 + p0[4]*w11, c0[5]*w10 + p0[5]*w11,
              c0[6]*w10 + p0[6]*w11, c0[7]*w10 + p0[7]*w11 };
    *(v4f*)(ob + nb0)     = a;
    *(v4f*)(ob + nb0 + 4) = b;

    const int nb1 = 16 + 8 * half;
    v4f c = { c1[0]*w10 + p1[0]*w11, c1[1]*w10 + p1[1]*w11,
              c1[2]*w10 + p1[2]*w11, c1[3]*w10 + p1[3]*w11 };
    *(v4f*)(ob + nb1) = c;
    if (half == 0) {
        v4f d = { c1[4]*w10 + p1[4]*w11, c1[5]*w10 + p1[5]*w11,
                  c1[6]*w10 + p1[6]*w11, c1[7]*w10 + p1[7]*w11 };
        *(v4f*)(ob + nb1 + 4) = d;
    }
}

__global__ __launch_bounds__(256)
void conv_gemm_roll_fused(const float* __restrict__ x,    // (1024, 64*28, 28)
                          const float* __restrict__ w1,   // (128, 2)
                          const float* __restrict__ w2,   // (64, 3, 128) == (192,128)
                          float* __restrict__ out)        // (1024, 128, 28, 28)
{
    __shared__ float    xs[2][J_ * H_];    // double-buffered x slice (14 KB)
    __shared__ _Float16 As[32 * AROW];     // im2col tile, f16        (13 KB)

    const int m    = blockIdx.x;
    const int tid  = threadIdx.x;
    const int lane = tid & 31;
    const int wid  = tid >> 5;            // 0..7 -> i-block
    const int half = lane >> 4;           // 0/1  (wmma lane half)
    const int lm   = lane & 15;
    const int i    = wid * 16 + lm;       // this lane's output channel

    // per-lane w1 scales: out = w10*t4[l_out-1] + w11*t4[l_out-2]
    const float w10 = w1[2*i + 0];
    const float w11 = w1[2*i + 1];

    const float* xm = x   + (size_t)m * (J_ * L_) * H_;
    float*       ob = out + (size_t)m * I_ * L_ * H_ + (size_t)i * (L_ * H_);

    // kick off slice 0 while we set up weights
    stage_slice(xm, 0, xs[0], tid);

    // ---- B fragments: w2 as (192,128) f16, resident in VGPRs for the whole kernel ----
    // B 32x16 in contiguous k order, same as A: lane half h holds K 16h..16h+15; element e -> K = half*16 + e
    v16h Bf[6];
    #pragma unroll
    for (int kk = 0; kk < 6; ++kk) {
        #pragma unroll
        for (int e = 0; e < 16; ++e) {
            const int k = kk*32 + half*16 + e;
            Bf[kk][e] = (_Float16)w2[k * I_ + i];
        }
    }

    // zero A tile once (pad rows 28..31 stay zero forever)
    for (int idx = tid; idx < 32 * AROW; idx += 256)
        As[idx] = (_Float16)0.f;

    v8f prev0 = {}, prev1 = {}, first0 = {}, first1 = {};

    for (int l = 0; l < L_; ++l) {
        wait_async_all();     // my chunks of slice l have landed in LDS
        __syncthreads();      // everyone's chunks landed; As readers of l-1 done

        // overlap: start DMA of slice l+1 into the other buffer
        if (l + 1 < L_) stage_slice(xm, l + 1, xs[(l + 1) & 1], tid);

        const float* cur = xs[l & 1];

        // im2col: As[n][j*3+k] = cur[j][n+k-1], zero at H boundaries
        for (int f = tid; f < J_ * H_; f += 256) {
            const int j = f / H_, n = f - j * H_;
            const float cm  = cur[j * H_ + n];
            const float lft = (n > 0)      ? cur[j * H_ + n - 1] : 0.f;
            const float rgt = (n < H_ - 1) ? cur[j * H_ + n + 1] : 0.f;
            _Float16* ar = &As[n * AROW + j * 3];
            ar[0] = (_Float16)lft;
            ar[1] = (_Float16)cm;
            ar[2] = (_Float16)rgt;
        }
        __syncthreads();

        // ---- GEMM tile: t4_l[n, i..] = A(32x192) x B(192x16), f32 accum ----
        v8f a0 = {}, a1 = {};
        #pragma unroll
        for (int kk = 0; kk < 6; ++kk) {
            const int k0 = kk * 32 + half * 16; // e -> K=k0+e
            {
                const _Float16* r = &As[lm * AROW + k0];          // rows 0..15
                v8h lo = *(const v8h*)(r);
                v8h hi = *(const v8h*)(r + 8);
                v16h af = __builtin_shufflevector(lo, hi,
                    0,1,2,3,4,5,6,7,8,9,10,11,12,13,14,15);
                a0 = __builtin_amdgcn_wmma_f32_16x16x32_f16(
                        false, af, false, Bf[kk], (short)0, a0, false, false);
            }
            {
                const _Float16* r = &As[(lm + 16) * AROW + k0];   // rows 16..31
                v8h lo = *(const v8h*)(r);
                v8h hi = *(const v8h*)(r + 8);
                v16h af = __builtin_shufflevector(lo, hi,
                    0,1,2,3,4,5,6,7,8,9,10,11,12,13,14,15);
                a1 = __builtin_amdgcn_wmma_f32_16x16x32_f16(
                        false, af, false, Bf[kk], (short)0, a1, false, false);
            }
        }

        if (l == 0) {
            first0 = a0; first1 = a1;       // t4_0, needed for l_out == 1
        } else {
            int lo_ = l + 1; if (lo_ >= L_) lo_ -= L_;   // out[(l+1)%28]
            emit_slice(ob + lo_ * H_, half, a0, prev0, a1, prev1, w10, w11);
        }
        prev0 = a0; prev1 = a1;
    }

    // l_out == 1 : w10 * t4_0 + w11 * t4_27
    emit_slice(ob + 1 * H_, half, first0, prev0, first1, prev1, w10, w11);
}

extern "C" void kernel_launch(void* const* d_in, const int* in_sizes, int n_in,
                              void* d_out, int out_size, void* d_ws, size_t ws_size,
                              hipStream_t stream) {
    const float* x  = (const float*)d_in[0];   // (1024, 1792, 28) f32
    const float* w1 = (const float*)d_in[1];   // (128, 2)         f32
    const float* w2 = (const float*)d_in[2];   // (64, 3, 128)     f32
    float* out = (float*)d_out;                // (1024, 128, 28, 28) f32

    conv_gemm_roll_fused<<<dim3(M_), dim3(256), 0, stream>>>(x, w1, w2, out);
}
